// GATPerso_27565100106037
// MI455X (gfx1250) — matrix operand order repaired and checked by reading, not compiled, not run on hardware
//
#include <hip/hip_runtime.h>
#include <cstdint>
#include <cstddef>

// ---------------- problem constants (match reference) ----------------
#define N_NODES 50000
#define N_EDGES 800000
#define N_GRAPH 256
#define C_IN    128
#define C_HID   64
#define N_HEADS 2
#define C_OUT   768
#define ATT_SLOPE 0.2f
#define ACT_SLOPE 0.01f

#define CEILDIV(a, b) (((a) + (b) - 1) / (b))

typedef __attribute__((ext_vector_type(16))) _Float16 v16h;
typedef __attribute__((ext_vector_type(8)))  _Float16 v8h;
typedef __attribute__((ext_vector_type(8)))  float    v8f;

// ------------- monotonic int encoding for float atomicMax -------------
__device__ __forceinline__ unsigned f2mono(float f) {
  unsigned b = __float_as_uint(f);
  return b ^ ((unsigned)((int)b >> 31) | 0x80000000u);
}
__device__ __forceinline__ float mono2f(unsigned k) {
  unsigned b = (k & 0x80000000u) ? (k & 0x7fffffffu) : ~k;
  return __uint_as_float(b);
}

// ---------------------------- utilities -------------------------------
__global__ void fill_u32_k(unsigned* __restrict__ p, unsigned v, size_t n) {
  size_t i = (size_t)blockIdx.x * blockDim.x + threadIdx.x;
  if (i < n) p[i] = v;
}

__global__ void cvt_f32_f16_k(const float* __restrict__ s, _Float16* __restrict__ d, size_t n) {
  size_t i = (size_t)blockIdx.x * blockDim.x + threadIdx.x;
  if (i < n) d[i] = (_Float16)s[i];
}

// --------------------------- WMMA GEMM --------------------------------
// out[M,Nc] = A[M,K] @ W[Nc,K]^T + bias, then leaky(slope). slope=1 => identity.
// One wave32 computes a 16x16 output tile; K multiple of 32; M,Nc multiples of 16.
__global__ void wmma_gemm_nt_k(const _Float16* __restrict__ A,
                               const _Float16* __restrict__ W,
                               const float* __restrict__ bias,
                               float* __restrict__ out,
                               int M, int K, int Nc, float slope) {
  int wave = blockIdx.x * (blockDim.x >> 5) + ((int)threadIdx.x >> 5);
  int lane = (int)threadIdx.x & 31;
  int tilesN = Nc >> 4;
  int nTiles = (M >> 4) * tilesN;
  if (wave >= nTiles) return;              // wave-uniform guard, EXEC stays all-1s
  int tm = wave / tilesN;
  int tn = wave - tm * tilesN;

  int row = tm * 16 + (lane & 15);
  int col = tn * 16 + (lane & 15);
  bool hi = lane >= 16;

  // A: lane m <- X[row, k+(hi?8:0)..+7] and X[row, k+(hi?8:0)+16..+23]
  // B: lane n <- W[col, k+(hi?8:0)..+7] and W[col, k+(hi?8:0)+16..+23]
  const _Float16* pa = A + (size_t)row * K + (hi ? 8 : 0);
  const _Float16* pw = W + (size_t)col * K + (hi ? 8 : 0);

  v8f acc = {};
  for (int k = 0; k < K; k += 32) {
    v8h a0 = *(const v8h*)(pa + k);
    v8h a1 = *(const v8h*)(pa + k + 16);
    v8h b0 = *(const v8h*)(pw + k);
    v8h b1 = *(const v8h*)(pw + k + 16);
    v16h a, b;
#pragma unroll
    for (int i = 0; i < 8; ++i) { a[i] = a0[i]; a[8 + i] = a1[i]; b[i] = b0[i]; b[8 + i] = b1[i]; }
    acc = __builtin_amdgcn_wmma_f32_16x16x32_f16(false, a, false, b,
                                                 (short)0, acc, false, false);
  }

  float bv = bias ? bias[col] : 0.0f;
  int rbase = tm * 16 + (hi ? 8 : 0);
#pragma unroll
  for (int r = 0; r < 8; ++r) {
    float v = acc[r] + bv;
    v = v > 0.0f ? v : slope * v;
    out[(size_t)(rbase + r) * Nc + col] = v;
  }
}

// ----------------------- GATv2 edge kernels ---------------------------
// Stage 1: per-edge attention logits + segment max over dst.
__global__ void edge_logits_k(const float* __restrict__ xl, const float* __restrict__ xr,
                              const float* __restrict__ att,
                              const int* __restrict__ srcE, const int* __restrict__ dstE,
                              int H, float* __restrict__ ebuf, unsigned* __restrict__ emax) {
  int e = blockIdx.x * blockDim.x + threadIdx.x;
  const int ET = N_EDGES + N_NODES;
  if (e >= ET) return;
  int s = (e < N_EDGES) ? srcE[e] : (e - N_EDGES);
  int d = (e < N_EDGES) ? dstE[e] : (e - N_EDGES);
  for (int h = 0; h < H; ++h) {
    const float4* pl = (const float4*)(xl + ((size_t)s * H + h) * C_HID);
    const float4* pr = (const float4*)(xr + ((size_t)d * H + h) * C_HID);
    const float4* pa = (const float4*)(att + (size_t)h * C_HID);
    float acc = 0.0f;
#pragma unroll
    for (int i = 0; i < C_HID / 4; ++i) {
      float4 a = pl[i], b = pr[i], w = pa[i];
      float v;
      v = a.x + b.x; v = v > 0.0f ? v : ATT_SLOPE * v; acc += v * w.x;
      v = a.y + b.y; v = v > 0.0f ? v : ATT_SLOPE * v; acc += v * w.y;
      v = a.z + b.z; v = v > 0.0f ? v : ATT_SLOPE * v; acc += v * w.z;
      v = a.w + b.w; v = v > 0.0f ? v : ATT_SLOPE * v; acc += v * w.w;
    }
    ebuf[(size_t)e * H + h] = acc;
    atomicMax(&emax[(size_t)d * H + h], f2mono(acc));
  }
}

// Stage 2: ex = exp(e - max); segment sum of ex over dst. ebuf updated in place.
__global__ void edge_expsum_k(const int* __restrict__ dstE, int H,
                              float* __restrict__ ebuf, const unsigned* __restrict__ emax,
                              float* __restrict__ denom) {
  size_t t = (size_t)blockIdx.x * blockDim.x + threadIdx.x;
  const size_t total = (size_t)(N_EDGES + N_NODES) * H;
  if (t >= total) return;
  size_t e = t / H;
  int h = (int)(t - e * H);
  int d = (e < N_EDGES) ? dstE[e] : (int)(e - N_EDGES);
  float ex = __expf(ebuf[t] - mono2f(emax[(size_t)d * H + h]));
  ebuf[t] = ex;
  atomicAdd(&denom[(size_t)d * H + h], ex);
}

// Stage 3: out[dst] += xl[src] * alpha   (per edge, head, channel; coalesced atomics)
__global__ void edge_scatter_k(const float* __restrict__ xl, const float* __restrict__ ebuf,
                               const float* __restrict__ denom,
                               const int* __restrict__ srcE, const int* __restrict__ dstE,
                               int H, float* __restrict__ out) {
  size_t t = (size_t)blockIdx.x * blockDim.x + threadIdx.x;
  const size_t total = (size_t)(N_EDGES + N_NODES) * H * C_HID;
  if (t >= total) return;
  int c = (int)(t & (C_HID - 1));
  size_t eh = t >> 6;                   // C_HID == 64
  size_t e = eh / H;
  int h = (int)(eh - e * H);
  int s = (e < N_EDGES) ? srcE[e] : (int)(e - N_EDGES);
  int d = (e < N_EDGES) ? dstE[e] : (int)(e - N_EDGES);
  float alpha = ebuf[eh] / denom[(size_t)d * H + h];
  atomicAdd(&out[((size_t)d * H + h) * C_HID + c],
            xl[((size_t)s * H + h) * C_HID + c] * alpha);
}

// h = leaky(h + bias[c], slope)
__global__ void bias_act_k(float* __restrict__ x, const float* __restrict__ bias,
                           int HC, size_t n, float slope) {
  size_t i = (size_t)blockIdx.x * blockDim.x + threadIdx.x;
  if (i >= n) return;
  int c = (int)(i % HC);
  float v = x[i] + bias[c];
  x[i] = v > 0.0f ? v : slope * v;
}

// ----------------------- global mean pool -----------------------------
__global__ void pool_sum_k(const float* __restrict__ h2, const int* __restrict__ batch,
                           float* __restrict__ sums, float* __restrict__ cnt) {
  size_t t = (size_t)blockIdx.x * blockDim.x + threadIdx.x;
  const size_t total = (size_t)N_NODES * C_HID;
  if (t >= total) return;
  size_t n = t >> 6;
  int c = (int)(t & (C_HID - 1));
  int g = batch[n];
  atomicAdd(&sums[(size_t)g * C_HID + c], h2[t]);
  if (c == 0) atomicAdd(&cnt[g], 1.0f);
}

__global__ void pool_div_k(const float* __restrict__ sums, const float* __restrict__ cnt,
                           _Float16* __restrict__ pooled_h) {
  int t = blockIdx.x * blockDim.x + threadIdx.x;
  const int total = N_GRAPH * C_HID;
  if (t >= total) return;
  int g = t >> 6;
  pooled_h[t] = (_Float16)(sums[t] / fmaxf(cnt[g], 1.0f));
}

// =======================================================================
extern "C" void kernel_launch(void* const* d_in, const int* in_sizes, int n_in,
                              void* d_out, int out_size, void* d_ws, size_t ws_size,
                              hipStream_t stream) {
  (void)in_sizes; (void)n_in; (void)out_size; (void)ws_size;

  const float* x      = (const float*)d_in[0];
  const int*   ei     = (const int*)d_in[1];
  const int*   batch  = (const int*)d_in[2];
  const float* w1_l   = (const float*)d_in[3];
  const float* b1_l   = (const float*)d_in[4];
  const float* w1_r   = (const float*)d_in[5];
  const float* b1_r   = (const float*)d_in[6];
  const float* att1   = (const float*)d_in[7];
  const float* bias1  = (const float*)d_in[8];
  const float* w2_l   = (const float*)d_in[9];
  const float* b2_l   = (const float*)d_in[10];
  const float* w2_r   = (const float*)d_in[11];
  const float* b2_r   = (const float*)d_in[12];
  const float* att2   = (const float*)d_in[13];
  const float* bias2  = (const float*)d_in[14];
  const float* fc1_w  = (const float*)d_in[15];
  const float* fc1_b  = (const float*)d_in[16];
  const float* fc2_w  = (const float*)d_in[17];
  const float* fc2_b  = (const float*)d_in[18];
  float* out = (float*)d_out;

  const int* srcE = ei;                 // edge_index[0]
  const int* dstE = ei + N_EDGES;       // edge_index[1]
  const int ET = N_EDGES + N_NODES;
  const int HC1 = N_HEADS * C_HID;      // 128

  // ------------------- workspace carve-out (deterministic) -------------------
  char* wsp = (char*)d_ws;
  size_t off = 0;
  auto alloc = [&](size_t bytes) -> void* {
    void* p = wsp + off;
    off = (off + bytes + 255) & ~(size_t)255;
    return p;
  };
  _Float16* x_h    = (_Float16*)alloc((size_t)N_NODES * C_IN * 2);
  _Float16* w1l_h  = (_Float16*)alloc((size_t)HC1 * C_IN * 2);
  _Float16* w1r_h  = (_Float16*)alloc((size_t)HC1 * C_IN * 2);
  _Float16* w2l_h  = (_Float16*)alloc((size_t)C_HID * HC1 * 2);
  _Float16* w2r_h  = (_Float16*)alloc((size_t)C_HID * HC1 * 2);
  _Float16* fc1w_h = (_Float16*)alloc((size_t)C_HID * C_HID * 2);
  _Float16* fc2w_h = (_Float16*)alloc((size_t)C_OUT * C_HID * 2);
  float*    xl1    = (float*)alloc((size_t)N_NODES * HC1 * 4);
  float*    xr1    = (float*)alloc((size_t)N_NODES * HC1 * 4);
  float*    h1     = (float*)alloc((size_t)N_NODES * HC1 * 4);
  _Float16* h1_h   = (_Float16*)alloc((size_t)N_NODES * HC1 * 2);
  float*    xl2    = (float*)alloc((size_t)N_NODES * C_HID * 4);
  float*    xr2    = (float*)alloc((size_t)N_NODES * C_HID * 4);
  float*    h2     = (float*)alloc((size_t)N_NODES * C_HID * 4);
  float*    ebuf   = (float*)alloc((size_t)ET * N_HEADS * 4);      // logits -> ex, reused by conv2
  unsigned* emax   = (unsigned*)alloc((size_t)N_NODES * N_HEADS * 4);
  float*    denom  = (float*)alloc((size_t)N_NODES * N_HEADS * 4);
  float*    sums   = (float*)alloc((size_t)N_GRAPH * C_HID * 4);
  float*    cnt    = (float*)alloc((size_t)N_GRAPH * 4);
  _Float16* pooled_h = (_Float16*)alloc((size_t)N_GRAPH * C_HID * 2);
  float*    fc1o   = (float*)alloc((size_t)N_GRAPH * C_HID * 4);
  _Float16* fc1o_h = (_Float16*)alloc((size_t)N_GRAPH * C_HID * 2);

  const int TB = 256;

  // ------------------- f32 -> f16 conversions -------------------
  {
    size_t n = (size_t)N_NODES * C_IN;
    cvt_f32_f16_k<<<CEILDIV(n, (size_t)TB), TB, 0, stream>>>(x, x_h, n);
  }
  cvt_f32_f16_k<<<CEILDIV(HC1 * C_IN, TB), TB, 0, stream>>>(w1_l, w1l_h, (size_t)HC1 * C_IN);
  cvt_f32_f16_k<<<CEILDIV(HC1 * C_IN, TB), TB, 0, stream>>>(w1_r, w1r_h, (size_t)HC1 * C_IN);
  cvt_f32_f16_k<<<CEILDIV(C_HID * HC1, TB), TB, 0, stream>>>(w2_l, w2l_h, (size_t)C_HID * HC1);
  cvt_f32_f16_k<<<CEILDIV(C_HID * HC1, TB), TB, 0, stream>>>(w2_r, w2r_h, (size_t)C_HID * HC1);
  cvt_f32_f16_k<<<CEILDIV(C_HID * C_HID, TB), TB, 0, stream>>>(fc1_w, fc1w_h, (size_t)C_HID * C_HID);
  cvt_f32_f16_k<<<CEILDIV(C_OUT * C_HID, TB), TB, 0, stream>>>(fc2_w, fc2w_h, (size_t)C_OUT * C_HID);

  // ------------------- conv1: linear transforms (WMMA) -------------------
  {
    int tiles = (N_NODES / 16) * (HC1 / 16);      // 3125 * 8
    int blocks = CEILDIV(tiles, TB / 32);
    wmma_gemm_nt_k<<<blocks, TB, 0, stream>>>(x_h, w1l_h, b1_l, xl1, N_NODES, C_IN, HC1, 1.0f);
    wmma_gemm_nt_k<<<blocks, TB, 0, stream>>>(x_h, w1r_h, b1_r, xr1, N_NODES, C_IN, HC1, 1.0f);
  }

  // ------------------- conv1: edge softmax + aggregate -------------------
  fill_u32_k<<<CEILDIV((size_t)N_NODES * N_HEADS, (size_t)TB), TB, 0, stream>>>(emax, 0u, (size_t)N_NODES * N_HEADS);
  fill_u32_k<<<CEILDIV((size_t)N_NODES * N_HEADS, (size_t)TB), TB, 0, stream>>>((unsigned*)denom, 0u, (size_t)N_NODES * N_HEADS);
  fill_u32_k<<<CEILDIV((size_t)N_NODES * HC1, (size_t)TB), TB, 0, stream>>>((unsigned*)h1, 0u, (size_t)N_NODES * HC1);

  edge_logits_k<<<CEILDIV(ET, TB), TB, 0, stream>>>(xl1, xr1, att1, srcE, dstE, N_HEADS, ebuf, emax);
  edge_expsum_k<<<CEILDIV((size_t)ET * N_HEADS, (size_t)TB), TB, 0, stream>>>(dstE, N_HEADS, ebuf, emax, denom);
  edge_scatter_k<<<CEILDIV((size_t)ET * N_HEADS * C_HID, (size_t)TB), TB, 0, stream>>>(xl1, ebuf, denom, srcE, dstE, N_HEADS, h1);
  bias_act_k<<<CEILDIV((size_t)N_NODES * HC1, (size_t)TB), TB, 0, stream>>>(h1, bias1, HC1, (size_t)N_NODES * HC1, ACT_SLOPE);

  // ------------------- conv2: linear transforms (WMMA) -------------------
  {
    size_t n = (size_t)N_NODES * HC1;
    cvt_f32_f16_k<<<CEILDIV(n, (size_t)TB), TB, 0, stream>>>(h1, h1_h, n);
    int tiles = (N_NODES / 16) * (C_HID / 16);    // 3125 * 4
    int blocks = CEILDIV(tiles, TB / 32);
    wmma_gemm_nt_k<<<blocks, TB, 0, stream>>>(h1_h, w2l_h, b2_l, xl2, N_NODES, HC1, C_HID, 1.0f);
    wmma_gemm_nt_k<<<blocks, TB, 0, stream>>>(h1_h, w2r_h, b2_r, xr2, N_NODES, HC1, C_HID, 1.0f);
  }

  // ------------------- conv2: edge softmax + aggregate (H=1) -------------------
  fill_u32_k<<<CEILDIV(N_NODES, TB), TB, 0, stream>>>(emax, 0u, (size_t)N_NODES);
  fill_u32_k<<<CEILDIV(N_NODES, TB), TB, 0, stream>>>((unsigned*)denom, 0u, (size_t)N_NODES);
  fill_u32_k<<<CEILDIV((size_t)N_NODES * C_HID, (size_t)TB), TB, 0, stream>>>((unsigned*)h2, 0u, (size_t)N_NODES * C_HID);

  edge_logits_k<<<CEILDIV(ET, TB), TB, 0, stream>>>(xl2, xr2, att2, srcE, dstE, 1, ebuf, emax);
  edge_expsum_k<<<CEILDIV((size_t)ET, (size_t)TB), TB, 0, stream>>>(dstE, 1, ebuf, emax, denom);
  edge_scatter_k<<<CEILDIV((size_t)ET * C_HID, (size_t)TB), TB, 0, stream>>>(xl2, ebuf, denom, srcE, dstE, 1, h2);
  bias_act_k<<<CEILDIV((size_t)N_NODES * C_HID, (size_t)TB), TB, 0, stream>>>(h2, bias2, C_HID, (size_t)N_NODES * C_HID, ACT_SLOPE);

  // ------------------- global mean pool -------------------
  fill_u32_k<<<CEILDIV(N_GRAPH * C_HID, TB), TB, 0, stream>>>((unsigned*)sums, 0u, (size_t)N_GRAPH * C_HID);
  fill_u32_k<<<1, TB, 0, stream>>>((unsigned*)cnt, 0u, (size_t)N_GRAPH);
  pool_sum_k<<<CEILDIV((size_t)N_NODES * C_HID, (size_t)TB), TB, 0, stream>>>(h2, batch, sums, cnt);
  pool_div_k<<<CEILDIV(N_GRAPH * C_HID, TB), TB, 0, stream>>>(sums, cnt, pooled_h);

  // ------------------- MLP head (WMMA) -------------------
  {
    int tiles1 = (N_GRAPH / 16) * (C_HID / 16);   // 16 * 4
    wmma_gemm_nt_k<<<CEILDIV(tiles1, TB / 32), TB, 0, stream>>>(pooled_h, fc1w_h, fc1_b, fc1o,
                                                                N_GRAPH, C_HID, C_HID, ACT_SLOPE);
    size_t n = (size_t)N_GRAPH * C_HID;
    cvt_f32_f16_k<<<CEILDIV(n, (size_t)TB), TB, 0, stream>>>(fc1o, fc1o_h, n);
    int tiles2 = (N_GRAPH / 16) * (C_OUT / 16);   // 16 * 48
    wmma_gemm_nt_k<<<CEILDIV(tiles2, TB / 32), TB, 0, stream>>>(fc1o_h, fc2w_h, fc2_b, out,
                                                                N_GRAPH, C_HID, C_OUT, 1.0f);
  }
}
